// PerformerSelfAttentionWrapper_59742995088073
// MI455X (gfx1250) — matrix operand order repaired and checked by reading, not compiled, not run on hardware
//
// Performer self-attention for MI455X (gfx1250), wave32 + WMMA bf16.
// v2.1: fix async builtin pointer types (AS1 int4* src, AS3 int4* dst),
//       vectorized (b128) fragment loads per CDNA5 WMMA VGPR layouts,
//       B operands staged/consumed in transposed (contiguous-per-lane) form.

#include <hip/hip_runtime.h>
#include <hip/hip_bf16.h>

typedef __bf16 bf16;
typedef __attribute__((ext_vector_type(16))) __bf16 v16bf;
typedef __attribute__((ext_vector_type(8)))  __bf16 v8bf;
typedef __attribute__((ext_vector_type(8)))  float  v8f;
typedef __attribute__((ext_vector_type(4)))  int    v4i;

#define B_    4
#define C_    256
#define H_    8
#define D_    32
#define M_    128
#define NPIX  9216
#define NTOT  (B_*NPIX)
#define NORM  0.42044820762685725f   /* 32^-0.25 */
#define RATIO 0.08838834764831845f   /* 128^-0.5 */
#define EPSF  1e-4f

__device__ __forceinline__ v8f wmma_bf16(v16bf a, v16bf b, v8f c) {
  return __builtin_amdgcn_wmma_f32_16x16x32_bf16(false, a, false, b, (short)0, c, false, false);
}

// ---- async global->LDS staging (CDNA5 ASYNCcnt path), guarded fallback ----
#if defined(__has_builtin)
#if __has_builtin(__builtin_amdgcn_global_load_async_to_lds_b128) && \
    __has_builtin(__builtin_amdgcn_s_wait_asynccnt)
#define ASYNC_LDS 1
#endif
#endif

__device__ __forceinline__ void cp16(const void* g, void* l) {
#ifdef ASYNC_LDS
  __builtin_amdgcn_global_load_async_to_lds_b128(
      (__attribute__((address_space(1))) v4i*)g,
      (__attribute__((address_space(3))) v4i*)l, 0, 0);
#else
  *(uint4*)l = *(const uint4*)g;
#endif
}
__device__ __forceinline__ void cp_wait() {
#ifdef ASYNC_LDS
  __builtin_amdgcn_s_wait_asynccnt(0);
#endif
}

// A tile 16x32 row-major (ld elems). Lane L<16: row L, K=0..7 & 16..23;
// lane L+16: row L, K=8..15 & 24..31. Two contiguous 16B chunks -> 2x ds_load_b128.
__device__ __forceinline__ v16bf frag_a(const bf16* base, int ld, int lane) {
  int row = lane & 15, kb = (lane >> 4) << 3;
  const bf16* p = base + row * ld + kb;
  v8bf lo = *(const v8bf*)p;
  v8bf hi = *(const v8bf*)(p + 16);
  v16bf f;
#pragma unroll
  for (int j = 0; j < 8; ++j) { f[j] = lo[j]; f[8 + j] = hi[j]; }
  return f;
}

// B tile 32x16 consumed from TRANSPOSED storage T[col][k] (ld elems).
// Lane L<16: col L, K=0..7 & 16..23; lane L+16: col L, K=8..15 & 24..31.
__device__ __forceinline__ v16bf frag_bT(const bf16* baseT, int ld, int lane) {
  int col = lane & 15, kb = (lane >> 4) << 3;
  const bf16* p = baseT + col * ld + kb;
  v8bf lo = *(const v8bf*)p;
  v8bf hi = *(const v8bf*)(p + 16);
  v16bf f;
#pragma unroll
  for (int j = 0; j < 8; ++j) { f[j] = lo[j]; f[8 + j] = hi[j]; }
  return f;
}

// ---------------- prep kernels ----------------

__global__ void k_cvt(const float* wq, const float* wk, const float* wv, const float* proj,
                      bf16* wqb, bf16* wkb, bf16* wvb, bf16* projb) {
  int i = blockIdx.x * 256 + threadIdx.x;
  if (i < 65536)        wqb[i]            = (bf16)wq[i];
  else if (i < 131072)  wkb[i - 65536]    = (bf16)wk[i - 65536];
  else if (i < 196608)  wvb[i - 131072]   = (bf16)wv[i - 131072];
  else if (i < 200704)  projb[i - 196608] = (bf16)proj[i - 196608];
}

__global__ void k_prep(const float* wp, const float* wo, const float* bo, const float* bp,
                       bf16* w2b, float* b2) {
  int o = blockIdx.x, c = threadIdx.x;
  float acc = 0.f;
  for (int k = 0; k < 256; ++k) acc += wp[o*256 + k] * wo[k*256 + c];
  w2b[o*256 + c] = (bf16)acc;
  if (c == 0) {
    float s = 0.f;
    for (int k = 0; k < 256; ++k) s += wp[o*256 + k] * bo[k];
    b2[o] = s + bp[o];
  }
}

__global__ void k_zero(float* ctx, float* ksum) {
  int i = blockIdx.x * 256 + threadIdx.x;
  if (i < 32*128*32) ctx[i] = 0.f;
  if (i < 32*128)    ksum[i] = 0.f;
}

__global__ void k_xf(const float* x, bf16* xf) {
  __shared__ float tile[32][33];
  int b = blockIdx.z, cb = blockIdx.y * 32, pb = blockIdx.x * 32;
  int tx = threadIdx.x, ty = threadIdx.y;
  for (int i = ty; i < 32; i += 8)
    tile[i][tx] = x[((size_t)(b*C_ + cb + i)) * NPIX + pb + tx];
  __syncthreads();
  for (int i = ty; i < 32; i += 8)
    xf[((size_t)(b*NPIX + pb + i)) * C_ + cb + tx] = (bf16)tile[tx][i];
}

// ---------------- QKV projection GEMM ----------------
// grid (NPIX/64, 3, B); block 256 (8 waves). Output (b,h,n,32) bf16.
__global__ void __launch_bounds__(256) k_qkv(const bf16* xf,
                                             const bf16* wqb, const bf16* wkb, const bf16* wvb,
                                             bf16* Qb, bf16* Kb, bf16* Vb) {
  __shared__ bf16 a_s[64*32];      // 4 KB  (row-major A chunk)
  __shared__ bf16 b_sT[256*32];    // 16 KB (b_sT[j][k] = W[j][kk*32+k] : B^T, natural copy)
  int b = blockIdx.z, wsel = blockIdx.y, nbase = blockIdx.x * 64;
  const bf16* W   = (wsel == 0) ? wqb : ((wsel == 1) ? wkb : wvb);
  bf16*       Out = (wsel == 0) ? Qb  : ((wsel == 1) ? Kb  : Vb);
  int tid = threadIdx.x, lane = tid & 31, wv = tid >> 5;
  int rt = wv & 3, cg = wv >> 2;
  v8f acc[8] = {};
  const bf16* xrow = xf + ((size_t)(b*NPIX + nbase)) * C_;
  for (int kk = 0; kk < 8; ++kk) {
    // A: thread t copies 16B chunk (row tid>>2, quarter tid&3)
    cp16(xrow + (size_t)(tid >> 2) * C_ + kk*32 + (tid & 3)*8, a_s + tid*8);
    // B^T: thread t copies 64B of weight row t
    {
      const bf16* wrow = W + (size_t)tid * 256 + kk*32;
#pragma unroll
      for (int p2 = 0; p2 < 4; ++p2) cp16(wrow + p2*8, b_sT + tid*32 + p2*8);
    }
    cp_wait();
    __syncthreads();
    if (kk < 7) __builtin_prefetch(xrow + (kk+1)*32, 0, 1);  // global_prefetch_b8
    v16bf a = frag_a(a_s + rt*16*32, 32, lane);
#pragma unroll
    for (int ct = 0; ct < 8; ++ct) {
      v16bf bb = frag_bT(b_sT + (cg*8 + ct)*16*32, 32, lane);
      acc[ct] = wmma_bf16(a, bb, acc[ct]);
    }
    __syncthreads();
  }
  int colh = lane & 15, rb = (lane >> 4) << 3;
#pragma unroll
  for (int ct = 0; ct < 8; ++ct) {
    int j = cg*128 + ct*16 + colh;
    int h = j >> 5, d = j & 31;
    bf16* dst = Out + (((size_t)(b*H_ + h)) * NPIX + nbase) * 32 + d;
#pragma unroll
    for (int g = 0; g < 8; ++g) {
      int r = rt*16 + g + rb;
      dst[(size_t)r * 32] = (bf16)acc[ct][g];
    }
  }
}

// ---------------- global key max (recompute pass, max-only) ----------------
// grid (NPIX/128, 32); block 256. proj is already B^T layout ([m][d]).
__global__ void __launch_bounds__(256) k_kmax(const bf16* Kb, const bf16* projb, float* kpart) {
  __shared__ bf16 a_in[128*32];  // 8 KB
  __shared__ bf16 pbuf[128*32];  // 8 KB, natural copy of proj[m][d]
  __shared__ float red[256];
  int bh = blockIdx.y, nb = blockIdx.x * 128;
  int tid = threadIdx.x, lane = tid & 31, wv = tid >> 5;
  const bf16* src = Kb + ((size_t)bh * NPIX + nb) * 32;
  for (int i = tid; i < 512; i += 256) cp16(src + i*8, a_in + i*8);
  for (int i = tid; i < 512; i += 256) cp16(projb + i*8, pbuf + i*8);
  cp_wait();
  __syncthreads();
  v16bf a = frag_a(a_in + wv*16*32, 32, lane);
  float lmax = -3.4e38f;
#pragma unroll
  for (int ct = 0; ct < 8; ++ct) {
    v8f accv = {};
    v16bf bb = frag_bT(pbuf + ct*16*32, 32, lane);
    accv = wmma_bf16(a, bb, accv);
#pragma unroll
    for (int g = 0; g < 8; ++g) lmax = fmaxf(lmax, accv[g]);
  }
  red[tid] = lmax;
  __syncthreads();
  for (int s = 128; s > 0; s >>= 1) { if (tid < s) red[tid] = fmaxf(red[tid], red[tid+s]); __syncthreads(); }
  if (tid == 0) kpart[bh*72 + blockIdx.x] = red[0] * NORM;
}

__global__ void k_kmax_red(const float* kpart, float* kmax) {
  __shared__ float red[128];
  int bh = blockIdx.x, t = threadIdx.x;
  red[t] = (t < 72) ? kpart[bh*72 + t] : -3.4e38f;
  __syncthreads();
  for (int s = 64; s > 0; s >>= 1) { if (t < s) red[t] = fmaxf(red[t], red[t+s]); __syncthreads(); }
  if (t == 0) kmax[bh] = red[0];
}

// ---------------- key features fused with ctx GEMM ----------------
// grid (NPIX/64, 32); block 256. ctx(128x32) += kp_tile^T @ v_tile; ksum += col sums.
__global__ void __launch_bounds__(256) k_kfeat_ctx(const bf16* Kb, const bf16* Vb, const bf16* projb,
                                                   const float* kmax, float* ctx, float* ksum) {
  __shared__ __align__(16) unsigned char smem[61440];
  bf16*  a_in = (bf16*)smem;              // 4 KB   [0,4096)
  bf16*  pbuf = (bf16*)(smem + 4096);     // 8 KB   [4096,12288)   proj natural (B^T)
  float* dash = (float*)(smem + 12288);   // 32 KB  [12288,45056)
  bf16*  kp_t = (bf16*)(smem + 45056);    // 16 KB  [45056,61440)  kp^T : [m][n']
  bf16*  vT   = (bf16*)(smem + 12288);    // 4 KB overlay on dash: v^T [d][n']
  int bh = blockIdx.y, nb = blockIdx.x * 64;
  int tid = threadIdx.x, lane = tid & 31, wv = tid >> 5;
  const bf16* src = Kb + ((size_t)bh * NPIX + nb) * 32;
  { // stage K rows + proj (async b128)
    cp16(src + tid*8, a_in + tid*8);
    for (int i = tid; i < 512; i += 256) cp16(projb + i*8, pbuf + i*8);
    cp_wait();
  }
  __syncthreads();
  { // dash = NORM * (k @ proj^T)
    int rt = wv & 3, cg = wv >> 2;
    int colh = lane & 15, rb = (lane >> 4) << 3;
    v16bf a = frag_a(a_in + rt*16*32, 32, lane);
#pragma unroll
    for (int ct = 0; ct < 4; ++ct) {
      v8f accv = {};
      v16bf bb = frag_bT(pbuf + (cg*4 + ct)*16*32, 32, lane);
      accv = wmma_bf16(a, bb, accv);
#pragma unroll
      for (int g = 0; g < 8; ++g)
        dash[(rt*16 + g + rb)*128 + (cg*4 + ct)*16 + colh] = accv[g] * NORM;
    }
  }
  __syncthreads();
  if (tid < 64) {  // exp in f32; write kp TRANSPOSED [m][n'] for vector A-frags
    float sq = 0.f;
#pragma unroll
    for (int d = 0; d < 32; ++d) { float kv = (float)a_in[tid*32 + d]; sq += kv*kv; }
    float diag = 0.5f * NORM * NORM * sq;
    float mx = kmax[bh];
    for (int m = 0; m < 128; ++m)
      kp_t[m*64 + tid] = (bf16)(RATIO * (__expf(dash[tid*128 + m] - diag - mx) + EPSF));
  }
  __syncthreads();   // dash dead; overlay region becomes vT
  { // stage v transposed: vT[d][n'] = v[n'][d]
    const bf16* vsrc = Vb + ((size_t)bh * NPIX + nb) * 32;
    for (int i = tid; i < 2048; i += 256) {
      int d = i >> 6, np = i & 63;
      vT[i] = vsrc[(size_t)np * 32 + d];
    }
    if (tid < 128) { // k_sum partials (rows of kp_t are contiguous)
      float s = 0.f;
      for (int r = 0; r < 64; ++r) s += (float)kp_t[tid*64 + r];
      atomicAdd(&ksum[bh*128 + tid], s);
    }
  }
  __syncthreads();
  { // ctx += kp^T @ v : A = kp_t rows (m), B^T = vT rows (d). 8 m-tiles x 2 d-tiles.
    v8f acc0 = {}, acc1 = {};
#pragma unroll
    for (int kc = 0; kc < 2; ++kc) {
      v16bf aT = frag_a(kp_t + wv*16*64 + kc*32, 64, lane);
      v16bf b0 = frag_bT(vT +         kc*32, 64, lane);
      v16bf b1 = frag_bT(vT + 16*64 + kc*32, 64, lane);
      acc0 = wmma_bf16(aT, b0, acc0);
      acc1 = wmma_bf16(aT, b1, acc1);
    }
    int colh = lane & 15, rb = (lane >> 4) << 3;
    float* cb = ctx + (size_t)bh * 128 * 32;
#pragma unroll
    for (int g = 0; g < 8; ++g) {
      int m = wv*16 + g + rb;
      atomicAdd(&cb[m*32 + colh],      acc0[g]);
      atomicAdd(&cb[m*32 + 16 + colh], acc1[g]);
    }
  }
}

// ---------------- extended context, stored TRANSPOSED (col-major) ----------------
// ctxEbT[bh][c][m], c in [0,48): c<32 -> ctx, c==32 -> k_sum, else 0.
__global__ void k_ctxE(const float* ctx, const float* ksum, bf16* ctxEbT) {
  int bh = blockIdx.x;
  for (int i = threadIdx.x; i < 48*128; i += 256) {
    int c = i >> 7, m = i & 127;
    float v = (c < 32) ? ctx[bh*4096 + m*32 + c] : ((c == 32) ? ksum[bh*128 + m] : 0.f);
    ctxEbT[(size_t)bh * 6144 + i] = (bf16)v;
  }
}

// ---------------- query features fused with output GEMM ----------------
// grid (NPIX/64, 32); block 256. out(64x48) = qp(64x128) @ ctxE(128x48); col 32 = denom.
__global__ void __launch_bounds__(256) k_qfeat_out(const bf16* Qb, const bf16* projb,
                                                   const bf16* ctxEbT, bf16* attn) {
  __shared__ __align__(16) unsigned char smem[61440];
  bf16*  a_in = (bf16*)smem;              // 4 KB   [0,4096)
  bf16*  pbuf = (bf16*)(smem + 4096);     // 8 KB   [4096,12288)
  float* dash = (float*)(smem + 12288);   // 32 KB  [12288,45056)
  bf16*  qp_l = (bf16*)(smem + 45056);    // 16 KB  [45056,61440)  qp row-major [n'][m]
  bf16*  ctxT = (bf16*)(smem + 12288);    // overlay: 12 KB  ctxE^T [c][m]
  float* outb = (float*)(smem + 24576);   // overlay: 12 KB  out f32 [n'][48]
  int bh = blockIdx.y, nb = blockIdx.x * 64;
  int tid = threadIdx.x, lane = tid & 31, wv = tid >> 5;
  const bf16* src = Qb + ((size_t)bh * NPIX + nb) * 32;
  {
    cp16(src + tid*8, a_in + tid*8);
    for (int i = tid; i < 512; i += 256) cp16(projb + i*8, pbuf + i*8);
    cp_wait();
  }
  __syncthreads();
  { // dash = NORM * (q @ proj^T)
    int rt = wv & 3, cg = wv >> 2;
    int colh = lane & 15, rb = (lane >> 4) << 3;
    v16bf a = frag_a(a_in + rt*16*32, 32, lane);
#pragma unroll
    for (int ct = 0; ct < 4; ++ct) {
      v8f accv = {};
      v16bf bb = frag_bT(pbuf + (cg*4 + ct)*16*32, 32, lane);
      accv = wmma_bf16(a, bb, accv);
#pragma unroll
      for (int g = 0; g < 8; ++g)
        dash[(rt*16 + g + rb)*128 + (cg*4 + ct)*16 + colh] = accv[g] * NORM;
    }
  }
  __syncthreads();
  if (tid < 64) {  // per-row max (query branch), exp in f32
    float mx = -3.4e38f;
    for (int m = 0; m < 128; ++m) mx = fmaxf(mx, dash[tid*128 + m]);
    float sq = 0.f;
#pragma unroll
    for (int d = 0; d < 32; ++d) { float qv = (float)a_in[tid*32 + d]; sq += qv*qv; }
    float diag = 0.5f * NORM * NORM * sq;
    for (int m = 0; m < 128; ++m)
      qp_l[tid*128 + m] = (bf16)(RATIO * (__expf(dash[tid*128 + m] - diag - mx) + EPSF));
  }
  __syncthreads();               // dash dead; overlay becomes ctxT/outb
  for (int i = tid; i < 768; i += 256) cp16(ctxEbT + (size_t)bh*6144 + i*8, ctxT + i*8);
  cp_wait();
  __syncthreads();
  if (wv < 4) {  // out = qp @ ctxE  (4 row-tiles x 3 col-tiles)
    v8f acc[3] = {};
#pragma unroll
    for (int kc = 0; kc < 4; ++kc) {
      v16bf a = frag_a(qp_l + wv*16*128 + kc*32, 128, lane);
#pragma unroll
      for (int ct = 0; ct < 3; ++ct) {
        v16bf bb = frag_bT(ctxT + ct*16*128 + kc*32, 128, lane);
        acc[ct] = wmma_bf16(a, bb, acc[ct]);
      }
    }
    int colh = lane & 15, rb = (lane >> 4) << 3;
#pragma unroll
    for (int ct = 0; ct < 3; ++ct)
#pragma unroll
      for (int g = 0; g < 8; ++g)
        outb[(wv*16 + g + rb)*48 + ct*16 + colh] = acc[ct][g];
  }
  __syncthreads();
  if (tid < 64) {  // scale by 1/denom, merge heads -> attn (b,n,256)
    float dinv = 1.f / outb[tid*48 + 32];
    int b = bh >> 3, h = bh & 7;
    bf16* dst = attn + ((size_t)b * NPIX + nb + tid) * 256 + h*32;
    for (int d = 0; d < 32; ++d) dst[d] = (bf16)(outb[tid*48 + d] * dinv);
  }
}

// ---------------- final fused GEMM (wo + 1x1 conv) with NCHW transpose ----------------
// grid (NPIX/64, B); block 256.
__global__ void __launch_bounds__(256) k_final(const bf16* attn, const bf16* w2b,
                                               const float* b2, float* out) {
  __shared__ bf16 a_s[64*32];
  __shared__ bf16 b_sT[256*32];
  int b = blockIdx.y, nbase = blockIdx.x * 64;
  int tid = threadIdx.x, lane = tid & 31, wv = tid >> 5;
  int rt = wv & 3, cg = wv >> 2;
  v8f acc[8] = {};
  const bf16* arow = attn + ((size_t)(b*NPIX + nbase)) * 256;
  for (int kk = 0; kk < 8; ++kk) {
    cp16(arow + (size_t)(tid >> 2) * 256 + kk*32 + (tid & 3)*8, a_s + tid*8);
    {
      const bf16* wrow = w2b + (size_t)tid * 256 + kk*32;
#pragma unroll
      for (int p2 = 0; p2 < 4; ++p2) cp16(wrow + p2*8, b_sT + tid*32 + p2*8);
    }
    cp_wait();
    __syncthreads();
    if (kk < 7) __builtin_prefetch(arow + (kk+1)*32, 0, 1);
    v16bf a = frag_a(a_s + rt*16*32, 32, lane);
#pragma unroll
    for (int ct = 0; ct < 8; ++ct) {
      v16bf bb = frag_bT(b_sT + (cg*8 + ct)*16*32, 32, lane);
      acc[ct] = wmma_bf16(a, bb, acc[ct]);
    }
    __syncthreads();
  }
  int colh = lane & 15, rb = (lane >> 4) << 3;
#pragma unroll
  for (int ct = 0; ct < 8; ++ct) {
    int j = cg*128 + ct*16 + colh;
    float bias = b2[j];
    float* dst = out + ((size_t)(b*C_ + j)) * NPIX + nbase;
#pragma unroll
    for (int g = 0; g < 8; ++g) {
      int r = rt*16 + g + rb;
      dst[r] = acc[ct][g] + bias;
    }
  }
}

extern "C" void kernel_launch(void* const* d_in, const int* in_sizes, int n_in,
                              void* d_out, int out_size, void* d_ws, size_t ws_size,
                              hipStream_t stream) {
  const float* x    = (const float*)d_in[0];
  const float* wq   = (const float*)d_in[1];
  const float* wk   = (const float*)d_in[2];
  const float* wv   = (const float*)d_in[3];
  const float* wo   = (const float*)d_in[4];
  const float* bo   = (const float*)d_in[5];
  const float* proj = (const float*)d_in[6];
  const float* wp   = (const float*)d_in[7];
  const float* bp   = (const float*)d_in[8];
  float* out = (float*)d_out;

  char* ws = (char*)d_ws;
  size_t off = 0;
  auto carve = [&](size_t bytes) -> char* {
    char* p = ws + off;
    off = (off + bytes + 255) & ~(size_t)255;
    return p;
  };
  bf16*  xf    = (bf16*)carve((size_t)NTOT * C_ * 2);
  bf16*  wqb   = (bf16*)carve(65536 * 2);
  bf16*  wkb   = (bf16*)carve(65536 * 2);
  bf16*  wvb   = (bf16*)carve(65536 * 2);
  bf16*  w2b   = (bf16*)carve(65536 * 2);
  bf16*  projb = (bf16*)carve(4096 * 2);
  float* b2    = (float*)carve(256 * 4);
  bf16*  Qb    = (bf16*)carve((size_t)B_*H_*NPIX*32 * 2);
  bf16*  Kb    = (bf16*)carve((size_t)B_*H_*NPIX*32 * 2);
  bf16*  Vb    = (bf16*)carve((size_t)B_*H_*NPIX*32 * 2);
  float* ctx   = (float*)carve(32 * 128 * 32 * 4);
  float* ksum  = (float*)carve(32 * 128 * 4);
  bf16*  ctxEbT= (bf16*)carve(32 * 48 * 128 * 2);
  float* kpart = (float*)carve(32 * 72 * 4);
  float* kmax  = (float*)carve(32 * 4);
  bf16*  attn  = (bf16*)carve((size_t)NTOT * C_ * 2);

  k_cvt      <<<784, 256, 0, stream>>>(wq, wk, wv, proj, wqb, wkb, wvb, projb);
  k_prep     <<<256, 256, 0, stream>>>(wp, wo, bo, bp, w2b, b2);
  k_zero     <<<512, 256, 0, stream>>>(ctx, ksum);
  k_xf       <<<dim3(NPIX/32, C_/32, B_), dim3(32, 8), 0, stream>>>(x, xf);
  k_qkv      <<<dim3(NPIX/64, 3, B_), 256, 0, stream>>>(xf, wqb, wkb, wvb, Qb, Kb, Vb);
  k_kmax     <<<dim3(NPIX/128, 32), 256, 0, stream>>>(Kb, projb, kpart);
  k_kmax_red <<<32, 128, 0, stream>>>(kpart, kmax);
  k_kfeat_ctx<<<dim3(NPIX/64, 32), 256, 0, stream>>>(Kb, Vb, projb, kmax, ctx, ksum);
  k_ctxE     <<<32, 256, 0, stream>>>(ctx, ksum, ctxEbT);
  k_qfeat_out<<<dim3(NPIX/64, 32), 256, 0, stream>>>(Qb, projb, ctxEbT, attn);
  k_final    <<<dim3(NPIX/64, B_), 256, 0, stream>>>(attn, w2b, b2, out);
}
